// BitwiseMLP_274877907496
// MI455X (gfx1250) — matrix operand order repaired and checked by reading, not compiled, not run on hardware
//
#include <hip/hip_runtime.h>
#include <hip/hip_bf16.h>

typedef __bf16 bf16_t;
typedef __bf16 bf16x4  __attribute__((ext_vector_type(4)));
typedef __bf16 bf16x8  __attribute__((ext_vector_type(8)));
typedef __bf16 bf16x16 __attribute__((ext_vector_type(16)));
typedef float  f32x8   __attribute__((ext_vector_type(8)));
typedef __attribute__((address_space(3))) bf16_t lds_bf16;

#define BM 128
#define BN 128
#define BK 64
#define LDSS 72   // padded LDS stride in bf16 elems (144B: 16B-aligned, bank-spread)

// ---------------------------------------------------------------------------
// CDNA5 async global->LDS copy (16B per lane, ASYNCcnt-tracked) + waits
// ---------------------------------------------------------------------------
__device__ __forceinline__ void async_copy_b128(unsigned lds_addr,
                                                const bf16_t* gptr) {
  asm volatile("global_load_async_to_lds_b128 %0, %1, off"
               :: "v"(lds_addr), "v"((unsigned long long)(uintptr_t)gptr)
               : "memory");
}
#define S_WAIT_ASYNCCNT(imm) asm volatile("s_wait_asynccnt " imm ::: "memory")

__device__ __forceinline__ unsigned lds_addr_of(const bf16_t* p) {
  return (unsigned)(unsigned long long)(lds_bf16*)p;
}

// ---------------------------------------------------------------------------
// f32 -> bf16 elementwise convert (vectorized x4)
// ---------------------------------------------------------------------------
__global__ void cvt_f32_to_bf16(const float* __restrict__ src,
                                bf16_t* __restrict__ dst, int n4) {
  int i = blockIdx.x * blockDim.x + threadIdx.x;
  if (i >= n4) return;
  float4 v = *(const float4*)(src + (size_t)i * 4);
  bf16x4 o;
  o.x = (bf16_t)v.x; o.y = (bf16_t)v.y; o.z = (bf16_t)v.z; o.w = (bf16_t)v.w;
  *(bf16x4*)(dst + (size_t)i * 4) = o;
}

// ---------------------------------------------------------------------------
// GEMM: Y[M,N] = A[M,K] (bf16, row-major) * W[N,K]^T (bf16, row-major) + bias
// 128x128 block tile, BK=64, 8 wave32 waves (2Mx4N), double-buffered LDS
// filled by global_load_async_to_lds_b128, compute = v_wmma_f32_16x16x32_bf16.
// ---------------------------------------------------------------------------
__global__ __launch_bounds__(256)
void gemm_bf16_nt(const bf16_t* __restrict__ A, const bf16_t* __restrict__ W,
                  const float* __restrict__ bias, float* __restrict__ Y,
                  int M, int N, int K) {
  __shared__ bf16_t lA[2][BM * LDSS];
  __shared__ bf16_t lW[2][BN * LDSS];

  const int tid  = threadIdx.x;
  const int lane = tid & 31;
  const int wave = tid >> 5;
  const int waveM = wave & 1;   // 2 waves along M
  const int waveN = wave >> 1;  // 4 waves along N
  const int m0 = blockIdx.y * BM;
  const int n0 = blockIdx.x * BN;

  // tile loader coordinates: each thread moves 64B (4 x b128) per matrix tile
  const int lrow = tid >> 1;            // 0..127
  const int lcol = (tid & 1) * 32;      // 0 or 32 (bf16 elems)

  const int hl  = lane >> 4;            // lane half (ISA VGPR layout)
  const int l16 = lane & 15;

  // per-thread staging addresses
  unsigned ldsA[2], ldsW[2];
  ldsA[0] = lds_addr_of(&lA[0][lrow * LDSS + lcol]);
  ldsA[1] = lds_addr_of(&lA[1][lrow * LDSS + lcol]);
  ldsW[0] = lds_addr_of(&lW[0][lrow * LDSS + lcol]);
  ldsW[1] = lds_addr_of(&lW[1][lrow * LDSS + lcol]);
  const bf16_t* gA = A + (size_t)(m0 + lrow) * K + lcol;
  const bf16_t* gW = W + (size_t)(n0 + lrow) * K + lcol;

  auto issue_tile = [&](int t, int buf) {
    const bf16_t* pa = gA + (size_t)t * BK;
    const bf16_t* pw = gW + (size_t)t * BK;
#pragma unroll
    for (int c = 0; c < 4; ++c) {          // 4 x 16B per matrix
      async_copy_b128(ldsA[buf] + c * 16, pa + c * 8);
      async_copy_b128(ldsW[buf] + c * 16, pw + c * 8);
    }
  };

  f32x8 acc[4][2];
#pragma unroll
  for (int i = 0; i < 4; ++i)
#pragma unroll
    for (int j = 0; j < 2; ++j)
      acc[i][j] = (f32x8){0.f, 0.f, 0.f, 0.f, 0.f, 0.f, 0.f, 0.f};

  const int T = K / BK;
  issue_tile(0, 0);

  for (int t = 0; t < T; ++t) {
    const int cur = t & 1;
    if (t + 1 < T) {
      issue_tile(t + 1, cur ^ 1);          // overlap next fill with compute
      S_WAIT_ASYNCCNT("0x8");              // only next tile's 8 copies pending
    } else {
      S_WAIT_ASYNCCNT("0x0");
    }
    __syncthreads();                       // all waves' fills of buf[cur] visible

#pragma unroll
    for (int kk = 0; kk < 2; ++kk) {       // two 16x16x32 K-steps per tile
      // B fragments (32x16): lane=N; lanes 0-15 hold K 0..15, 16-31 hold 16..31
      union { bf16x16 f; bf16x8 h[2]; } bfrag[2];
#pragma unroll
      for (int j = 0; j < 2; ++j) {
        const int n  = waveN * 32 + j * 16 + l16;
        const int kb = kk * 32 + hl * 16;
        bfrag[j].h[0] = *(const bf16x8*)(&lW[cur][n * LDSS + kb]);
        bfrag[j].h[1] = *(const bf16x8*)(&lW[cur][n * LDSS + kb + 8]);
      }
#pragma unroll
      for (int i = 0; i < 4; ++i) {
        // A fragment (16x32): lanes 0-15: K 0..15; 16-31: K 16..31
        union { bf16x16 f; bf16x8 h[2]; } afrag;
        const int m  = waveM * 64 + i * 16 + l16;
        const int ka = kk * 32 + hl * 16;
        afrag.h[0] = *(const bf16x8*)(&lA[cur][m * LDSS + ka]);
        afrag.h[1] = *(const bf16x8*)(&lA[cur][m * LDSS + ka + 8]);
#pragma unroll
        for (int j = 0; j < 2; ++j) {
          acc[i][j] = __builtin_amdgcn_wmma_f32_16x16x32_bf16(
              false, afrag.f, false, bfrag[j].f, (short)0, acc[i][j], false, false);
        }
      }
    }
    __syncthreads();   // all reads of buf[cur] done before t+1 refills it
  }

  // epilogue: C/D layout -> lane&15 = N; VGPR r + 8*(lane>>4) = M
#pragma unroll
  for (int j = 0; j < 2; ++j) {
    const int n = n0 + waveN * 32 + j * 16 + l16;
    const float bv = bias[n];
#pragma unroll
    for (int i = 0; i < 4; ++i) {
      const int mb = m0 + waveM * 64 + i * 16 + hl * 8;
#pragma unroll
      for (int r = 0; r < 8; ++r) {
        Y[(size_t)(mb + r) * N + n] = acc[i][j][r] + bv;
      }
    }
  }
}

// ---------------------------------------------------------------------------
// Column partial sums: chunk of rows per blockIdx.y, one column per thread
// ---------------------------------------------------------------------------
__global__ void col_partial(const float* __restrict__ Y, float* __restrict__ psum,
                            float* __restrict__ psq, int N, int rowsPerChunk) {
  const int col = blockIdx.x * blockDim.x + threadIdx.x;
  const int chunk = blockIdx.y;
  const size_t r0 = (size_t)chunk * rowsPerChunk;
  float s = 0.f, q = 0.f;
  for (int r = 0; r < rowsPerChunk; ++r) {
    float v = Y[(r0 + r) * N + col];
    s += v;
    q += v * v;
  }
  psum[chunk * N + col] = s;
  psq[chunk * N + col]  = q;
}

__global__ void col_finalize(const float* __restrict__ psum, const float* __restrict__ psq,
                             float* __restrict__ mean, float* __restrict__ inv,
                             int N, int chunks, float invM) {
  const int col = blockIdx.x * blockDim.x + threadIdx.x;
  float s = 0.f, q = 0.f;
  for (int c = 0; c < chunks; ++c) {
    s += psum[c * N + col];
    q += psq[c * N + col];
  }
  const float m = s * invM;
  const float v = q * invM - m * m;
  mean[col] = m;
  inv[col]  = rsqrtf(v + 1e-5f);
}

// ---------------------------------------------------------------------------
// Apply BN: out = gamma*(y-mean)*inv + beta. dstF (f32) and/or dstB (bf16).
// ---------------------------------------------------------------------------
__global__ void bn_apply(const float* __restrict__ Y, const float* __restrict__ mean,
                         const float* __restrict__ inv, const float* __restrict__ gamma,
                         const float* __restrict__ beta, float* __restrict__ dstF,
                         bf16_t* __restrict__ dstB, int N, size_t total) {
  const size_t i = (size_t)blockIdx.x * blockDim.x + threadIdx.x;
  if (i >= total) return;
  const int col = (int)(i % N);
  const float o = gamma[col] * (Y[i] - mean[col]) * inv[col] + beta[col];
  if (dstF) dstF[i] = o;
  if (dstB) dstB[i] = (bf16_t)o;
}

// ---------------------------------------------------------------------------
extern "C" void kernel_launch(void* const* d_in, const int* in_sizes, int n_in,
                              void* d_out, int out_size, void* d_ws, size_t ws_size,
                              hipStream_t stream) {
  const float* x   = (const float*)d_in[0];
  const float* W0  = (const float*)d_in[1];
  const float* b0  = (const float*)d_in[2];
  const float* g0  = (const float*)d_in[3];
  const float* be0 = (const float*)d_in[4];
  const float* W1  = (const float*)d_in[5];
  const float* b1  = (const float*)d_in[6];
  const float* g1  = (const float*)d_in[7];
  const float* be1 = (const float*)d_in[8];
  const float* W2  = (const float*)d_in[9];
  const float* b2  = (const float*)d_in[10];
  const float* g2  = (const float*)d_in[11];
  const float* be2 = (const float*)d_in[12];

  const int B = 16384, D0 = 1024, DH = 2048, DOUT = 1024;
  const int CHUNKS = 32, ROWS_PER_CHUNK = B / CHUNKS;

  char* ws = (char*)d_ws;
  size_t off = 0;
  auto alloc = [&](size_t bytes) -> char* {
    char* p = ws + off;
    off = (off + bytes + 255) & ~(size_t)255;
    return p;
  };
  bf16_t* Abuf = (bf16_t*)alloc((size_t)B * DH * sizeof(bf16_t));   // ping
  bf16_t* Hbuf = (bf16_t*)alloc((size_t)B * DH * sizeof(bf16_t));   // pong
  float*  Ybuf = (float*)alloc((size_t)B * DH * sizeof(float));
  bf16_t* W0b  = (bf16_t*)alloc((size_t)DH * D0 * sizeof(bf16_t));
  bf16_t* W1b  = (bf16_t*)alloc((size_t)DH * DH * sizeof(bf16_t));
  bf16_t* W2b  = (bf16_t*)alloc((size_t)DOUT * DH * sizeof(bf16_t));
  float*  psum = (float*)alloc((size_t)CHUNKS * DH * sizeof(float));
  float*  psq  = (float*)alloc((size_t)CHUNKS * DH * sizeof(float));
  float*  mean = (float*)alloc((size_t)DH * sizeof(float));
  float*  inv  = (float*)alloc((size_t)DH * sizeof(float));

  auto cvt = [&](const float* src, bf16_t* dst, size_t n) {
    int n4 = (int)(n / 4);
    cvt_f32_to_bf16<<<(n4 + 255) / 256, 256, 0, stream>>>(src, dst, n4);
  };

  // f32 -> bf16 for activations-in and all weights
  cvt(x,  Abuf, (size_t)B * D0);
  cvt(W0, W0b,  (size_t)DH * D0);
  cvt(W1, W1b,  (size_t)DH * DH);
  cvt(W2, W2b,  (size_t)DOUT * DH);

  auto layer = [&](const bf16_t* Ain, const bf16_t* Wb, const float* bias,
                   const float* gamma, const float* beta, int N, int K,
                   float* dstF, bf16_t* dstB) {
    dim3 grid(N / BN, B / BM);
    gemm_bf16_nt<<<grid, 256, 0, stream>>>(Ain, Wb, bias, Ybuf, B, N, K);
    dim3 sgrid(N / 256, CHUNKS);
    col_partial<<<sgrid, 256, 0, stream>>>(Ybuf, psum, psq, N, ROWS_PER_CHUNK);
    col_finalize<<<N / 256, 256, 0, stream>>>(psum, psq, mean, inv, N, CHUNKS,
                                              1.0f / (float)B);
    size_t total = (size_t)B * N;
    bn_apply<<<(unsigned)((total + 255) / 256), 256, 0, stream>>>(
        Ybuf, mean, inv, gamma, beta, dstF, dstB, N, total);
  };

  // Layer 0: [B,1024] x [2048,1024]^T -> bf16 Hbuf
  layer(Abuf, W0b, b0, g0, be0, DH, D0, nullptr, Hbuf);
  // Layer 1: [B,2048] x [2048,2048]^T -> bf16 Abuf
  layer(Hbuf, W1b, b1, g1, be1, DH, DH, nullptr, Abuf);
  // Layer 2: [B,2048] x [1024,2048]^T -> f32 d_out
  layer(Abuf, W2b, b2, g2, be2, DOUT, DH, (float*)d_out, nullptr);
}
